// SportsGraphNeuralNetwork_37838661878106
// MI455X (gfx1250) — matrix operand order repaired and checked by reading, not compiled, not run on hardware
//
#include <hip/hip_runtime.h>
#include <hip/hip_bf16.h>
#include <cmath>

// ---------------------------------------------------------------------------
// SportsGraphNeuralNetwork (3x GATConv + mean-pool + MLP) for MI455X (gfx1250)
// GEMMs via v_wmma_f32_16x16x32_bf16 (wave32, 16x16 C tile per wave), with
// A pre-converted to bf16 once per layer so the WMMA loop is pure load+mma.
// Edge softmax/aggregation via f32 global atomics (L2-resident working set).
// ---------------------------------------------------------------------------

#define NODES   20000
#define DIN     128
#define HIDC    128      // per-head channels
#define NHEADS  4
#define HCH     (NHEADS * HIDC)   // 512

typedef __attribute__((ext_vector_type(16))) __bf16 v16bf;
typedef __attribute__((ext_vector_type(8)))  __bf16 v8bf;
typedef __attribute__((ext_vector_type(8)))  float  v8f;

static inline int cdiv_i(long long a, int b) { return (int)((a + b - 1) / b); }
static inline size_t align256(size_t x) { return (x + 255) & ~(size_t)255; }

// ---------------------------------------------------------------- utilities
__global__ void fill_f32_k(float* __restrict__ p, float v, int n) {
  int i = blockIdx.x * blockDim.x + threadIdx.x;
  if (i < n) p[i] = v;
}

// elementwise f32 -> bf16 (A operand staging; amortized over 32 column tiles)
__global__ void bf16cast_k(const float* __restrict__ src,
                           __bf16* __restrict__ dst, int n) {
  int i = blockIdx.x * blockDim.x + threadIdx.x;
  if (i < n) dst[i] = (__bf16)src[i];
}

// W [K x N] f32 row-major  ->  Wt [N x K] bf16 row-major (contiguous B frags)
__global__ void transpose_to_bf16_k(const float* __restrict__ W,
                                    __bf16* __restrict__ Wt, int K, int N) {
  int i = blockIdx.x * blockDim.x + threadIdx.x;
  if (i >= K * N) return;
  int k = i / N, n = i % N;
  Wt[(size_t)n * K + k] = (__bf16)W[i];
}

// ------------------------------------------------------------------- GEMM
// C[M,N] = A[M,K](bf16) @ Bt[N,K](bf16)^T, f32 accumulate.
// One wave -> one 16x16 tile of C. 8 waves / block cover 128 columns; all 8
// waves share the same A rows (L1/L2 served).
// A fragment (16x32 bf16): lane m = lane&15, khalf = lane>>4;
//   elements e=0..7  -> k = k0 + 16*khalf + e        (contiguous 16B)
//   elements e=8..15 -> k = k0 + 16*khalf + 8 + e-8  (contiguous 16B)
// B fragment (32x16 bf16): lane n = lane&15, k = k0 + 16*khalf + e (32B)
__global__ __launch_bounds__(256) void gemm_bf16_wmma_k(
    const __bf16* __restrict__ A, const __bf16* __restrict__ Bt,
    float* __restrict__ C, int M, int N, int K) {
  const int lane = threadIdx.x & 31;
  const int wave = threadIdx.x >> 5;
  const int m16  = blockIdx.x * 16;
  const int n16  = (blockIdx.y * 8 + wave) * 16;
  if (n16 >= N) return;                       // wave-uniform
  const int half = lane >> 4;
  const int l15  = lane & 15;

  const __bf16* ap = A  + (size_t)(m16 + l15) * K + 16 * half;
  const __bf16* bp = Bt + (size_t)(n16 + l15) * K + 16 * half;

  v8f acc = {};
  for (int k0 = 0; k0 < K; k0 += 32) {
    // pull next K-slab toward L1 while this step's WMMA executes
    // (speculative global_prefetch_b8; OOB prefetch is silently dropped)
    __builtin_prefetch(ap + k0 + 64, 0, 0);
    __builtin_prefetch(bp + k0 + 64, 0, 0);
    v8bf alo = *(const v8bf*)(ap + k0);       // 16B load: e0..7
    v8bf ahi = *(const v8bf*)(ap + k0 + 8);   // 16B load: e8..15
    v16bf a;
#pragma unroll
    for (int i = 0; i < 8; ++i) {
      a[i]     = alo[i];
      a[i + 8] = ahi[i];
    }
    v16bf b = *(const v16bf*)(bp + k0);       // 32B contiguous bf16 load
    acc = __builtin_amdgcn_wmma_f32_16x16x32_bf16(
        /*neg_a=*/false, a, /*neg_b=*/false, b,
        /*c_mod=*/(short)0, acc, /*reuse_a=*/false, /*reuse_b=*/false);
  }
  // C/D layout: vgpr r, lanes 0-15 -> M=r, lanes 16-31 -> M=r+8; N = lane&15
  float* cp = C + (size_t)(m16 + 8 * half) * N + n16 + l15;
#pragma unroll
  for (int r = 0; r < 8; ++r) cp[(size_t)r * N] = acc[r];
}

// -------------------------------------------------- attention coefficients
// asrc[n,h] = dot(XL[n,h,:], a_s[h,:]);  adst likewise.  One wave per (n,h).
__global__ __launch_bounds__(256) void attn_coeff_k(
    const float* __restrict__ XL, const float* __restrict__ a_s,
    const float* __restrict__ a_d, float* __restrict__ asrc,
    float* __restrict__ adst, int Nn, int H, int Cch) {
  int gw   = (blockIdx.x * blockDim.x + threadIdx.x) >> 5;
  int lane = threadIdx.x & 31;
  if (gw >= Nn * H) return;                   // wave-uniform
  int n = gw / H, h = gw % H;
  const float* xp = XL + ((size_t)n * H + h) * Cch;
  float s = 0.f, d = 0.f;
  for (int c = lane; c < Cch; c += 32) {
    float v = xp[c];
    s += v * a_s[h * Cch + c];
    d += v * a_d[h * Cch + c];
  }
#pragma unroll
  for (int off = 16; off; off >>= 1) {
    s += __shfl_down(s, off, 32);
    d += __shfl_down(d, off, 32);
  }
  if (lane == 0) { asrc[gw] = s; adst[gw] = d; }
}

// ----------------------------------------------------------- edge kernels
__device__ __forceinline__ void edge_ends(const int* __restrict__ ei, int E,
                                          int e, int& s, int& d) {
  if (e < E) { s = ei[e]; d = ei[E + e]; }
  else       { s = d = e - E; }               // self loops appended
}

__device__ __forceinline__ void atomicMaxF(float* addr, float value) {
  // sign-aware int trick (works for buffers initialized to -inf)
  if (value >= 0.f) atomicMax((int*)addr, __float_as_int(value));
  else              atomicMin((unsigned int*)addr, __float_as_uint(value));
}

__device__ __forceinline__ float leaky02(float v) {
  return v >= 0.f ? v : 0.2f * v;
}

__global__ void edge_max_k(const int* __restrict__ ei, int E, int H,
                           const float* __restrict__ asrc,
                           const float* __restrict__ adst,
                           float* __restrict__ mx, int Etot) {
  int e = blockIdx.x * blockDim.x + threadIdx.x;
  if (e >= Etot) return;
  int s, d; edge_ends(ei, E, e, s, d);
  for (int h = 0; h < H; ++h) {
    float v = leaky02(asrc[s * H + h] + adst[d * H + h]);
    atomicMaxF(&mx[d * H + h], v);
  }
}

__global__ void edge_exp_k(const int* __restrict__ ei, int E, int H,
                           const float* __restrict__ asrc,
                           const float* __restrict__ adst,
                           const float* __restrict__ mx,
                           float* __restrict__ exb, float* __restrict__ den,
                           int Etot) {
  int e = blockIdx.x * blockDim.x + threadIdx.x;
  if (e >= Etot) return;
  int s, d; edge_ends(ei, E, e, s, d);
  for (int h = 0; h < H; ++h) {
    float v  = leaky02(asrc[s * H + h] + adst[d * H + h]);
    float ex = expf(v - mx[d * H + h]);
    exb[(size_t)e * H + h] = ex;
    atomicAdd(&den[d * H + h], ex);
  }
}

// out[dst, h, c] += (exb[e,h]/den[dst,h]) * XL[src, h, c]; flat edge x channel
__global__ void edge_agg_k(const int* __restrict__ ei, int E, int H, int Cch,
                           const float* __restrict__ exb,
                           const float* __restrict__ den,
                           const float* __restrict__ XL,
                           float* __restrict__ out, long long total) {
  long long idx = (long long)blockIdx.x * blockDim.x + threadIdx.x;
  if (idx >= total) return;
  int HC = H * Cch;
  int c  = (int)(idx % HC);
  int e  = (int)(idx / HC);
  int h  = c / Cch;
  int s, d; edge_ends(ei, E, e, s, d);
  float alpha = exb[(size_t)e * H + h] / den[d * H + h];
  atomicAdd(&out[(size_t)d * HC + c], alpha * XL[(size_t)s * HC + c]);
}

__global__ void bias_act_k(float* __restrict__ hbuf,
                           const float* __restrict__ bias, int total, int C,
                           int do_relu) {
  int i = blockIdx.x * blockDim.x + threadIdx.x;
  if (i >= total) return;
  float v = hbuf[i] + bias[i % C];
  hbuf[i] = do_relu ? fmaxf(v, 0.f) : v;
}

// ------------------------------------------------------------- pool + MLP
__global__ __launch_bounds__(256) void colmean_k(const float* __restrict__ h,
                                                 float* __restrict__ g, int Nn,
                                                 int C) {
  __shared__ float sd[256];
  int c = blockIdx.x;
  float s = 0.f;
  for (int n = threadIdx.x; n < Nn; n += blockDim.x) s += h[(size_t)n * C + c];
  sd[threadIdx.x] = s;
  __syncthreads();
  for (int off = 128; off; off >>= 1) {
    if (threadIdx.x < off) sd[threadIdx.x] += sd[threadIdx.x + off];
    __syncthreads();
  }
  if (threadIdx.x == 0) g[c] = sd[0] / (float)Nn;
}

__global__ __launch_bounds__(64) void mlp_k(const float* __restrict__ g,
                                            const float* __restrict__ Wm1,
                                            const float* __restrict__ bm1,
                                            const float* __restrict__ Wm2,
                                            const float* __restrict__ bm2,
                                            float* __restrict__ out, int C,
                                            int Hd) {
  __shared__ float hid[64];
  int j = threadIdx.x;
  if (j < Hd) {
    float s = bm1[j];
    for (int i = 0; i < C; ++i) s += g[i] * Wm1[i * Hd + j];
    hid[j] = fmaxf(s, 0.f);
  }
  __syncthreads();
  if (j == 0) {
    float s = bm2[0];
    for (int i = 0; i < Hd; ++i) s += hid[i] * Wm2[i];
    out[0] = s;
  }
}

// ---------------------------------------------------------------- driver
extern "C" void kernel_launch(void* const* d_in, const int* in_sizes, int n_in,
                              void* d_out, int out_size, void* d_ws,
                              size_t ws_size, hipStream_t stream) {
  (void)n_in; (void)out_size; (void)ws_size;
  const float* x   = (const float*)d_in[0];
  const int*   ei  = (const int*)d_in[1];
  const float* W0  = (const float*)d_in[2];
  const float* as0 = (const float*)d_in[3];
  const float* ad0 = (const float*)d_in[4];
  const float* b0  = (const float*)d_in[5];
  const float* W1  = (const float*)d_in[6];
  const float* as1 = (const float*)d_in[7];
  const float* ad1 = (const float*)d_in[8];
  const float* b1  = (const float*)d_in[9];
  const float* W2  = (const float*)d_in[10];
  const float* as2 = (const float*)d_in[11];
  const float* ad2 = (const float*)d_in[12];
  const float* b2  = (const float*)d_in[13];
  const float* Wm1 = (const float*)d_in[14];
  const float* bm1 = (const float*)d_in[15];
  const float* Wm2 = (const float*)d_in[16];
  const float* bm2 = (const float*)d_in[17];

  const int Nn = in_sizes[0] / DIN;       // 20000
  const int E  = in_sizes[1] / 2;         // 320000
  const int ET = E + Nn;                  // + self loops

  // ---- workspace carve ----
  char*  w    = (char*)d_ws;
  size_t off  = 0;
  auto carve  = [&](size_t bytes) {
    void* p = w + off;
    off += align256(bytes);
    return p;
  };
  float*  XL   = (float*)carve((size_t)Nn * HCH * sizeof(float));      // GEMM out
  float*  HB   = (float*)carve((size_t)Nn * HCH * sizeof(float));      // layer io
  float*  HF   = (float*)carve((size_t)Nn * HIDC * sizeof(float));     // layer-2 out
  __bf16* ABF  = (__bf16*)carve((size_t)Nn * HCH * sizeof(__bf16));    // A bf16
  __bf16* Wt   = (__bf16*)carve((size_t)HCH * HCH * sizeof(__bf16));   // W^T bf16
  float*  ASRC = (float*)carve((size_t)Nn * NHEADS * sizeof(float));
  float*  ADST = (float*)carve((size_t)Nn * NHEADS * sizeof(float));
  float*  MX   = (float*)carve((size_t)Nn * NHEADS * sizeof(float));
  float*  DEN  = (float*)carve((size_t)Nn * NHEADS * sizeof(float));
  float*  EXB  = (float*)carve((size_t)ET * NHEADS * sizeof(float));
  float*  G    = (float*)carve(HIDC * sizeof(float));

  const int TB = 256;
  auto run_layer = [&](const float* Ain, int K, int Nw, int H, int Cch,
                       const float* Wl, const float* a_s, const float* a_d,
                       const float* bias, float* aggout, int do_relu) {
    // 1) stage operands in bf16, then XL = Ain @ Wl (WMMA, f32 accumulate)
    bf16cast_k<<<cdiv_i((long long)Nn * K, TB), TB, 0, stream>>>(Ain, ABF,
                                                                 Nn * K);
    transpose_to_bf16_k<<<cdiv_i((long long)K * Nw, TB), TB, 0, stream>>>(
        Wl, Wt, K, Nw);
    gemm_bf16_wmma_k<<<dim3(Nn / 16, Nw / 128), TB, 0, stream>>>(
        ABF, Wt, XL, Nn, Nw, K);
    // 2) attention coefficients
    attn_coeff_k<<<cdiv_i((long long)Nn * H * 32, TB), TB, 0, stream>>>(
        XL, a_s, a_d, ASRC, ADST, Nn, H, Cch);
    // 3) segment softmax over incoming edges
    fill_f32_k<<<cdiv_i((long long)Nn * H, TB), TB, 0, stream>>>(
        MX, -INFINITY, Nn * H);
    fill_f32_k<<<cdiv_i((long long)Nn * H, TB), TB, 0, stream>>>(
        DEN, 0.f, Nn * H);
    edge_max_k<<<cdiv_i(ET, TB), TB, 0, stream>>>(ei, E, H, ASRC, ADST, MX, ET);
    edge_exp_k<<<cdiv_i(ET, TB), TB, 0, stream>>>(ei, E, H, ASRC, ADST, MX,
                                                  EXB, DEN, ET);
    // 4) weighted scatter-add aggregation
    fill_f32_k<<<cdiv_i((long long)Nn * H * Cch, TB), TB, 0, stream>>>(
        aggout, 0.f, Nn * H * Cch);
    long long total = (long long)ET * H * Cch;
    edge_agg_k<<<cdiv_i(total, TB), TB, 0, stream>>>(ei, E, H, Cch, EXB, DEN,
                                                     XL, aggout, total);
    // 5) bias (+ ReLU)
    bias_act_k<<<cdiv_i((long long)Nn * H * Cch, TB), TB, 0, stream>>>(
        aggout, bias, Nn * H * Cch, H * Cch, do_relu);
  };

  // layer 0: [N,128] -> [N,512], 4 heads concat, ReLU
  run_layer(x, DIN, HCH, NHEADS, HIDC, W0, as0, ad0, b0, HB, 1);
  // layer 1: [N,512] -> [N,512], 4 heads concat, ReLU   (HB in, HB out: input
  // is consumed by the bf16 staging pass before aggregation zero-fills it)
  run_layer(HB, HCH, HCH, NHEADS, HIDC, W1, as1, ad1, b1, HB, 1);
  // layer 2: [N,512] -> [N,128], 1 head, no activation
  run_layer(HB, HCH, HIDC, 1, HIDC, W2, as2, ad2, b2, HF, 0);

  // global mean pool + MLP head
  colmean_k<<<HIDC, 256, 0, stream>>>(HF, G, Nn, HIDC);
  mlp_k<<<1, 64, 0, stream>>>(G, Wm1, bm1, Wm2, bm2, (float*)d_out, HIDC,
                              HIDC / 2);
}
